// TreeLSTM_65403761983554
// MI455X (gfx1250) — compile-verified, matrix operand order checked by reading, not run on hardware
//
#include <hip/hip_runtime.h>

#define LL   16
#define NN   2048
#define CC   8
#define DD   64
#define HH   256
#define HIDN 256
#define FF   527
#define FPAD 544
#define G4   1024
#define B1   (LL*NN)   /* 32768 */
#define B2   (2*B1)    /* 65536 */

typedef __attribute__((ext_vector_type(16))) _Float16 v16h;
typedef __attribute__((ext_vector_type(8)))  _Float16 v8h;
typedef __attribute__((ext_vector_type(8)))  float    v8f;

union U16 { v16h v; v8h h[2]; };

// Load a 16x32 f16 A/B fragment from row-major [rows, ld] memory (global or LDS).
// Per-lane pattern (CDNA5 ISA 7.12.2): lanes 0-15 hold K 0..7 & 16..23,
// lanes 16-31 hold K 8..15 & 24..31, row = lane&15.
__device__ inline v16h load_frag16(const _Float16* p, int ld, int lane) {
  int r  = lane & 15;
  int kk = (lane >> 4) << 3;          // 0 or 8
  U16 u;
  u.h[0] = *(const v8h*)(p + (long)r * ld + kk);
  u.h[1] = *(const v8h*)(p + (long)r * ld + kk + 16);
  return u.v;
}

__device__ inline v8f wmma16(v16h a, v16h b, v8f c) {
  return __builtin_amdgcn_wmma_f32_16x16x32_f16(false, a, false, b, (short)0, c,
                                                false, false);
}

// C/D 16x16 f32 layout: VGPR r, lane l -> row r + (l>=16)*8, col l&15.
__device__ inline void store_acc(float* p, int ld, v8f a, int lane) {
  int col = lane & 15;
  int rb  = (lane >> 4) << 3;
#pragma unroll
  for (int r = 0; r < 8; ++r) p[(long)(rb + r) * ld + col] = a[r];
}

__device__ inline float sigf(float x) { return 1.f / (1.f + __expf(-x)); }

// ---------------------------------------------------------------- converts
__global__ void cvt_kernel(const float* s, _Float16* d, long n) {
  long i = (long)blockIdx.x * blockDim.x + threadIdx.x;
  if (i < n) d[i] = (_Float16)s[i];
}
__global__ void cvtpad_kernel(const float* s, _Float16* d, int rows, int sc, int dc) {
  long i = (long)blockIdx.x * blockDim.x + threadIdx.x;
  if (i < (long)rows * dc) {
    int r = (int)(i / dc), c = (int)(i % dc);
    d[i] = (c < sc) ? (_Float16)s[(long)r * sc + c] : (_Float16)0.f;
  }
}
__global__ void addvec_kernel(const float* a, const float* b, float* o, int n) {
  int i = blockIdx.x * blockDim.x + threadIdx.x;
  if (i < n) o[i] = a[i] + b[i];
}

// ---------------------------------------------------------------- lstm1
// One workgroup owns 32 batch rows, runs all 8 timesteps. 8 waves; wave w owns
// hidden columns [w*32, w*32+32) for ALL FOUR gates, so the whole cell update
// (sigmoid/tanh, c state) lives in WMMA accumulator registers; c persists in
// VGPRs across timesteps. Only h round-trips through double-buffered LDS
// (needed as next step's A fragments). Each B weight fragment feeds 2 WMMAs
// (two M tiles), halving L2 weight traffic vs an M=16 tiling.
__global__ __launch_bounds__(256) void lstm1_kernel(
    const float* cond1, const float* cond2,          // [B1, 8, 64] each
    const _Float16* Wih, const _Float16* Whh,        // [1024,64],[1024,256]
    const float* bsum,                               // [1024] (b_ih+b_hh)
    _Float16* hout)                                  // [B2, 256]
{
  __shared__ __align__(16) _Float16 lx[32 * 64];         // 4 KB
  __shared__ __align__(16) _Float16 lhbuf[2][32 * 256];  // 2 x 16 KB
  int  tid = threadIdx.x, lane = tid & 31, wave = tid >> 5;
  long b0  = (long)blockIdx.x * 32;
  int  col = lane & 15, rb = (lane >> 4) << 3;

  // bias per (gate, jtile) for this lane's column
  float bias[4][2];
#pragma unroll
  for (int g = 0; g < 4; ++g)
#pragma unroll
    for (int jt = 0; jt < 2; ++jt)
      bias[g][jt] = bsum[g * 256 + wave * 32 + jt * 16 + col];

  // persistent cell state in registers: [jtile][mtile][r]
  float creg[2][2][8];
#pragma unroll
  for (int a = 0; a < 2; ++a)
#pragma unroll
    for (int b = 0; b < 2; ++b)
#pragma unroll
      for (int r = 0; r < 8; ++r) creg[a][b][r] = 0.f;

  _Float16* lcur = lhbuf[0];
  _Float16* lnxt = lhbuf[1];
  for (int i = tid; i < 32 * 256; i += 256) lcur[i] = (_Float16)0.f;
  __syncthreads();

  for (int t = 0; t < CC; ++t) {
    // stage X_t [32 x 64] as f16 in LDS
    for (int i = tid; i < 32 * 64; i += 256) {
      int  r = i >> 6, k = i & 63;
      long b = b0 + r;
      const float* base = (b < B1) ? (cond1 + (b * CC + t) * DD)
                                   : (cond2 + ((b - B1) * CC + t) * DD);
      lx[r * 64 + k] = (_Float16)base[k];
    }
    __syncthreads();

#pragma unroll
    for (int jt = 0; jt < 2; ++jt) {
      int j0 = wave * 32 + jt * 16;      // hidden-column base of this tile
      __builtin_prefetch(Whh + (long)j0 * 256, 0, 1);
      v8f zero = {};
      v8f acc[4][2];
#pragma unroll
      for (int g = 0; g < 4; ++g) { acc[g][0] = zero; acc[g][1] = zero; }

      // x @ Wih^T contribution (K = 64)
#pragma unroll
      for (int k0 = 0; k0 < 64; k0 += 32) {
        v16h a0 = load_frag16(lx + k0, 64, lane);
        v16h a1 = load_frag16(lx + 16 * 64 + k0, 64, lane);
#pragma unroll
        for (int g = 0; g < 4; ++g) {
          v16h bb = load_frag16(Wih + (long)(g * 256 + j0) * 64 + k0, 64, lane);
          acc[g][0] = wmma16(a0, bb, acc[g][0]);
          acc[g][1] = wmma16(a1, bb, acc[g][1]);
        }
      }
      // h @ Whh^T contribution (K = 256)
#pragma unroll 2
      for (int k0 = 0; k0 < 256; k0 += 32) {
        v16h a0 = load_frag16(lcur + k0, 256, lane);
        v16h a1 = load_frag16(lcur + 16 * 256 + k0, 256, lane);
#pragma unroll
        for (int g = 0; g < 4; ++g) {
          v16h bb = load_frag16(Whh + (long)(g * 256 + j0) * 256 + k0, 256, lane);
          acc[g][0] = wmma16(a0, bb, acc[g][0]);
          acc[g][1] = wmma16(a1, bb, acc[g][1]);
        }
      }
      // cell update entirely in registers (gate order i, f, g, o)
#pragma unroll
      for (int mt = 0; mt < 2; ++mt) {
#pragma unroll
        for (int r = 0; r < 8; ++r) {
          float gi = acc[0][mt][r] + bias[0][jt];
          float gf = acc[1][mt][r] + bias[1][jt];
          float gg = acc[2][mt][r] + bias[2][jt];
          float go = acc[3][mt][r] + bias[3][jt];
          float c  = sigf(gf) * creg[jt][mt][r] + sigf(gi) * tanhf(gg);
          creg[jt][mt][r] = c;
          float h  = sigf(go) * tanhf(c);
          int row  = mt * 16 + rb + r;
          lnxt[row * 256 + j0 + col] = (_Float16)h;
        }
      }
    }
    __syncthreads();   // lx rewrite + lnxt becomes next step's read buffer
    _Float16* tmp = lcur; lcur = lnxt; lnxt = tmp;
  }

  for (int i = tid; i < 32 * 256; i += 256) {
    int r = i >> 8, j = i & 255;
    hout[(b0 + r) * 256 + j] = lcur[i];
  }
}

// ---------------------------------------------------------------- cond linear
// lo_avg = 0.5*(relu(h1 @ W^T + b) + relu(h2 @ W^T + b)), [B1, 256] f32
__global__ __launch_bounds__(128) void condlin_kernel(
    const _Float16* h1, const _Float16* W, const float* bias, float* lo)
{
  int  tid = threadIdx.x, lane = tid & 31, wave = tid >> 5;
  long s0  = (long)blockIdx.x * 16;
  for (int nt = 0; nt < 4; ++nt) {
    int n0 = wave * 64 + nt * 16;
    v8f a1 = {}, a2 = {};
#pragma unroll
    for (int k0 = 0; k0 < 256; k0 += 32) {
      v16h bb = load_frag16(W + (long)n0 * 256 + k0, 256, lane);
      v16h x1 = load_frag16(h1 + s0 * 256 + k0, 256, lane);
      v16h x2 = load_frag16(h1 + (s0 + B1) * 256 + k0, 256, lane);
      a1 = wmma16(x1, bb, a1);
      a2 = wmma16(x2, bb, a2);
    }
    int col = lane & 15, rb = (lane >> 4) << 3;
#pragma unroll
    for (int r = 0; r < 8; ++r) {
      float bv = bias[n0 + col];
      float y1 = fmaxf(a1[r] + bv, 0.f);
      float y2 = fmaxf(a2[r] + bv, 0.f);
      lo[(s0 + rb + r) * 256 + n0 + col] = 0.5f * (y1 + y2);
    }
  }
}

// ---------------------------------------------------------------- column stats
__global__ __launch_bounds__(256) void colstats_kernel(
    const float* x, int rows, int cols, float* mean, float* var)
{
  int   col = blockIdx.x;
  float s = 0.f, s2 = 0.f;
  for (int r = threadIdx.x; r < rows; r += 256) {
    float v = x[(long)r * cols + col];
    s += v; s2 += v * v;
  }
  __shared__ float sb[256], sc[256];
  sb[threadIdx.x] = s; sc[threadIdx.x] = s2;
  __syncthreads();
  for (int o = 128; o > 0; o >>= 1) {
    if (threadIdx.x < o) { sb[threadIdx.x] += sb[threadIdx.x + o];
                           sc[threadIdx.x] += sc[threadIdx.x + o]; }
    __syncthreads();
  }
  if (threadIdx.x == 0) {
    float m = sb[0] / rows;
    mean[col] = m;
    var[col]  = sc[0] / rows - m * m;
  }
}

// ---------------------------------------------------------------- sample enc
// so = relu(samples @ sampW^T + b) * mask -> feats cols [271, 527) as f16.
// The block's 16x1024 A tile is converted f32->f16 ONCE into LDS (zero-padded
// past K=1000); all four N tiles then consume clean ds_load_b128 fragments.
// B weights pre-converted to padded f16 [256,1024].
__global__ __launch_bounds__(128) void sample_kernel(
    const float* samples, const _Float16* Wsamp, const float* bsamp,
    const float* mask, _Float16* feats)
{
  __shared__ __align__(16) _Float16 la[16 * 1024];   // 32 KB
  int  tid = threadIdx.x, lane = tid & 31, wave = tid >> 5;
  long s0  = (long)blockIdx.x * 16;

  for (int i = tid; i < 16 * 1024; i += 128) {
    int r = i >> 10, k = i & 1023;
    la[i] = (k < 1000) ? (_Float16)samples[(s0 + r) * 1000 + k] : (_Float16)0.f;
  }
  __syncthreads();

  for (int nt = 0; nt < 4; ++nt) {
    int n0 = wave * 64 + nt * 16;
    v8f acc = {};
#pragma unroll 4
    for (int k0 = 0; k0 < 1024; k0 += 32) {
      v16h a  = load_frag16(la + k0, 1024, lane);
      v16h bb = load_frag16(Wsamp + (long)n0 * 1024 + k0, 1024, lane);
      acc = wmma16(a, bb, acc);
    }
    int col = lane & 15, rb = (lane >> 4) << 3;
#pragma unroll
    for (int r = 0; r < 8; ++r) {
      long srow = s0 + rb + r;
      float y = fmaxf(acc[r] + bsamp[n0 + col], 0.f) * mask[srow];
      feats[srow * FPAD + 271 + n0 + col] = (_Float16)y;
    }
  }
}

// ---------------------------------------------------------------- feats assembly
// cols [0,15): operators; [15,271): bn1(lo_avg); [271,527): from sample_kernel;
// [527,544): zero pad.
__global__ void feats_kernel(const float* ops, const float* lo,
                             const float* m, const float* v,
                             const float* g, const float* b, _Float16* feats)
{
  long i = (long)blockIdx.x * blockDim.x + threadIdx.x;
  if (i >= (long)B1 * FPAD) return;
  long row = i / FPAD; int col = (int)(i % FPAD);
  if (col < 15) {
    feats[i] = (_Float16)ops[row * 15 + col];
  } else if (col < 271) {
    int j = col - 15;
    float x = lo[row * 256 + j];
    feats[i] = (_Float16)((x - m[j]) * rsqrtf(v[j] + 1e-5f) * g[j] + b[j]);
  } else if (col >= 527) {
    feats[i] = (_Float16)0.f;
  }
}

// ---------------------------------------------------------------- lstm2 level
// One launch per tree level: gather-average children h/c, gates WMMA
// (K = 544 feats + 256 hidden), cell update. 8 waves x 128 gate cols.
__global__ __launch_bounds__(256) void lstm2_kernel(
    const _Float16* feats, int lev, int first, const int* mapping,
    const _Float16* hprev, const float* cprev,
    const _Float16* Wih, const _Float16* Whh, const float* bsum,
    _Float16* hnext16, float* hnext32, float* cnext)
{
  __shared__ __align__(16) _Float16 lh2[16 * 256];
  __shared__ __align__(16) float    lc2[16 * 256];
  __shared__ __align__(16) float    lg[16 * 1024];
  int tid = threadIdx.x, lane = tid & 31, wave = tid >> 5;
  int m0  = blockIdx.x * 16;

  for (int i = tid; i < 16 * 256; i += 256) {
    int r = i >> 8, j = i & 255;
    float hv = 0.f, cv = 0.f;
    if (!first) {
      int  n  = m0 + r;
      int  j0 = mapping[((long)lev * NN + n) * 2 + 0];
      int  j1 = mapping[((long)lev * NN + n) * 2 + 1];
      float h0 = j0 ? (float)hprev[(long)(j0 - 1) * 256 + j] : 0.f;
      float h1 = j1 ? (float)hprev[(long)(j1 - 1) * 256 + j] : 0.f;
      float c0 = j0 ? cprev[(long)(j0 - 1) * 256 + j] : 0.f;
      float c1 = j1 ? cprev[(long)(j1 - 1) * 256 + j] : 0.f;
      hv = 0.5f * (h0 + h1); cv = 0.5f * (c0 + c1);
    }
    lh2[i] = (_Float16)hv; lc2[i] = cv;
  }
  __syncthreads();

  const _Float16* fbase = feats + ((long)lev * NN + m0) * FPAD;
  for (int nt = 0; nt < 8; ++nt) {
    int n0 = wave * 128 + nt * 16;
    v8f acc = {};
#pragma unroll 4
    for (int k0 = 0; k0 < FPAD; k0 += 32) {          // feats @ Wih^T
      v16h a  = load_frag16(fbase + k0, FPAD, lane);
      v16h bb = load_frag16(Wih + (long)n0 * FPAD + k0, FPAD, lane);
      acc = wmma16(a, bb, acc);
    }
#pragma unroll
    for (int k0 = 0; k0 < 256; k0 += 32) {           // h2 @ Whh^T
      v16h a  = load_frag16(lh2 + k0, 256, lane);
      v16h bb = load_frag16(Whh + (long)n0 * 256 + k0, 256, lane);
      acc = wmma16(a, bb, acc);
    }
    store_acc(lg + n0, 1024, acc, lane);
  }
  __syncthreads();

  for (int i = tid; i < 16 * 256; i += 256) {
    int r = i >> 8, j = i & 255;
    float gi = lg[r * 1024 + j]       + bsum[j];
    float gf = lg[r * 1024 + 256 + j] + bsum[256 + j];
    float gg = lg[r * 1024 + 512 + j] + bsum[512 + j];
    float go = lg[r * 1024 + 768 + j] + bsum[768 + j];
    float c  = sigf(gf) * lc2[i] + sigf(gi) * tanhf(gg);
    float h  = sigf(go) * tanhf(c);
    long  o  = (long)(m0 + r) * 256 + j;
    cnext[o] = c; hnext16[o] = (_Float16)h; hnext32[o] = h;
  }
}

// ---------------------------------------------------------------- bn apply -> f16
__global__ void bnapply_kernel(const float* x, const float* m, const float* v,
                               const float* g, const float* b, _Float16* o,
                               long n, int cols)
{
  long i = (long)blockIdx.x * blockDim.x + threadIdx.x;
  if (i >= n) return;
  int j = (int)(i % cols);
  o[i] = (_Float16)((x[i] - m[j]) * rsqrtf(v[j] + 1e-5f) * g[j] + b[j]);
}

// ---------------------------------------------------------------- head GEMM
// out = relu(A[2048,256] @ W[256,256]^T + bias), f32
__global__ __launch_bounds__(128) void gemm_relu_kernel(
    const _Float16* A, const _Float16* W, const float* bias, float* out)
{
  int  tid = threadIdx.x, lane = tid & 31, wave = tid >> 5;
  long m0  = (long)blockIdx.x * 16;
  for (int nt = 0; nt < 4; ++nt) {
    int n0 = wave * 64 + nt * 16;
    v8f acc = {};
#pragma unroll
    for (int k0 = 0; k0 < 256; k0 += 32) {
      v16h a  = load_frag16(A + m0 * 256 + k0, 256, lane);
      v16h bb = load_frag16(W + (long)n0 * 256 + k0, 256, lane);
      acc = wmma16(a, bb, acc);
    }
    int col = lane & 15, rb = (lane >> 4) << 3;
#pragma unroll
    for (int r = 0; r < 8; ++r)
      out[(m0 + rb + r) * 256 + n0 + col] = fmaxf(acc[r] + bias[n0 + col], 0.f);
  }
}

// ---------------------------------------------------------------- head output
__global__ void headout_kernel(const float* y2, const float* Wo, const float* bo,
                               float* out, int off0, int off1, int off2)
{
  int i = blockIdx.x * blockDim.x + threadIdx.x;
  if (i >= NN) return;
  float s = bo[0];
  for (int k = 0; k < 256; ++k) s += y2[(long)i * 256 + k] * Wo[k];
  float o = 1.f / (1.f + __expf(-s));
  out[off0 + i] = o; out[off1 + i] = o; out[off2 + i] = o;
}

// ================================================================ host
extern "C" void kernel_launch(void* const* d_in, const int* in_sizes, int n_in,
                              void* d_out, int out_size, void* d_ws, size_t ws_size,
                              hipStream_t stream)
{
  (void)in_sizes; (void)n_in; (void)out_size; (void)ws_size;

  const float* operators = (const float*)d_in[0];
  const float* cond1     = (const float*)d_in[2];
  const float* cond2     = (const float*)d_in[3];
  const float* samples   = (const float*)d_in[4];
  const float* masks     = (const float*)d_in[5];
  const int*   mapping   = (const int*)  d_in[6];
  const float* W_ih1 = (const float*)d_in[7],  *W_hh1 = (const float*)d_in[8];
  const float* b_ih1 = (const float*)d_in[9],  *b_hh1 = (const float*)d_in[10];
  const float* W_ih2 = (const float*)d_in[11], *W_hh2 = (const float*)d_in[12];
  const float* b_ih2 = (const float*)d_in[13], *b_hh2 = (const float*)d_in[14];
  const float* cond_W = (const float*)d_in[15], *cond_b = (const float*)d_in[16];
  const float* samp_W = (const float*)d_in[17], *samp_b = (const float*)d_in[18];
  const float* bn1_g = (const float*)d_in[19], *bn1_b = (const float*)d_in[20];
  const float* bn2_g = (const float*)d_in[21], *bn2_b = (const float*)d_in[22];
  const float* bn3_g = (const float*)d_in[23], *bn3_b = (const float*)d_in[24];
  const float* t1_W2 = (const float*)d_in[25], *t1_b2 = (const float*)d_in[26];
  const float* t1_W3 = (const float*)d_in[27], *t1_b3 = (const float*)d_in[28];
  const float* t1_Wo = (const float*)d_in[29], *t1_bo = (const float*)d_in[30];
  const float* t2_W2 = (const float*)d_in[31], *t2_b2 = (const float*)d_in[32];
  const float* t2_W3 = (const float*)d_in[33], *t2_b3 = (const float*)d_in[34];
  const float* t2_Wo = (const float*)d_in[35], *t2_bo = (const float*)d_in[36];
  float* out = (float*)d_out;

  // workspace carving
  size_t off = 0;
  auto alloc = [&](size_t bytes) -> void* {
    off = (off + 255) & ~(size_t)255;
    void* p = (char*)d_ws + off;
    off += bytes;
    return p;
  };
  _Float16* wih1h  = (_Float16*)alloc((size_t)G4 * DD * 2);
  _Float16* whh1h  = (_Float16*)alloc((size_t)G4 * HH * 2);
  _Float16* wih2h  = (_Float16*)alloc((size_t)G4 * FPAD * 2);
  _Float16* whh2h  = (_Float16*)alloc((size_t)G4 * HH * 2);
  _Float16* condwh = (_Float16*)alloc((size_t)HIDN * HH * 2);
  _Float16* sampwh = (_Float16*)alloc((size_t)HIDN * 1024 * 2);
  _Float16* t1w2h  = (_Float16*)alloc((size_t)HIDN * HH * 2);
  _Float16* t1w3h  = (_Float16*)alloc((size_t)HIDN * HIDN * 2);
  _Float16* t2w2h  = (_Float16*)alloc((size_t)HIDN * HH * 2);
  _Float16* t2w3h  = (_Float16*)alloc((size_t)HIDN * HIDN * 2);
  float* bsum1 = (float*)alloc(G4 * 4);
  float* bsum2 = (float*)alloc(G4 * 4);
  _Float16* h1out = (_Float16*)alloc((size_t)B2 * HH * 2);
  float*    loavg = (float*)alloc((size_t)B1 * HIDN * 4);
  _Float16* feats = (_Float16*)alloc((size_t)B1 * FPAD * 2);
  float* m1 = (float*)alloc(256 * 4); float* v1 = (float*)alloc(256 * 4);
  float* m2 = (float*)alloc(256 * 4); float* v2 = (float*)alloc(256 * 4);
  float* m3 = (float*)alloc(256 * 4); float* v3 = (float*)alloc(256 * 4);
  _Float16* h16[2]; float* h32[2]; float* c32[2];
  for (int i = 0; i < 2; ++i) {
    h16[i] = (_Float16*)alloc((size_t)NN * HH * 2);
    h32[i] = (float*)alloc((size_t)NN * HH * 4);
    c32[i] = (float*)alloc((size_t)NN * HH * 4);
  }
  _Float16* zbuf = (_Float16*)alloc((size_t)NN * HH * 2);
  float*    y1   = (float*)alloc((size_t)NN * HIDN * 4);
  _Float16* y1n  = (_Float16*)alloc((size_t)NN * HIDN * 2);
  float*    y2   = (float*)alloc((size_t)NN * HIDN * 4);

  auto nb = [](long n, int t) { return (unsigned)((n + t - 1) / t); };

  // weight conversion + combined biases
  cvt_kernel<<<nb((long)G4 * DD, 256), 256, 0, stream>>>(W_ih1, wih1h, (long)G4 * DD);
  cvt_kernel<<<nb((long)G4 * HH, 256), 256, 0, stream>>>(W_hh1, whh1h, (long)G4 * HH);
  cvtpad_kernel<<<nb((long)G4 * FPAD, 256), 256, 0, stream>>>(W_ih2, wih2h, G4, FF, FPAD);
  cvt_kernel<<<nb((long)G4 * HH, 256), 256, 0, stream>>>(W_hh2, whh2h, (long)G4 * HH);
  cvt_kernel<<<nb((long)HIDN * HH, 256), 256, 0, stream>>>(cond_W, condwh, (long)HIDN * HH);
  cvtpad_kernel<<<nb((long)HIDN * 1024, 256), 256, 0, stream>>>(samp_W, sampwh, HIDN, 1000, 1024);
  cvt_kernel<<<nb(65536, 256), 256, 0, stream>>>(t1_W2, t1w2h, 65536);
  cvt_kernel<<<nb(65536, 256), 256, 0, stream>>>(t1_W3, t1w3h, 65536);
  cvt_kernel<<<nb(65536, 256), 256, 0, stream>>>(t2_W2, t2w2h, 65536);
  cvt_kernel<<<nb(65536, 256), 256, 0, stream>>>(t2_W3, t2w3h, 65536);
  addvec_kernel<<<nb(G4, 256), 256, 0, stream>>>(b_ih1, b_hh1, bsum1, G4);
  addvec_kernel<<<nb(G4, 256), 256, 0, stream>>>(b_ih2, b_hh2, bsum2, G4);

  // lstm1 over cond1+cond2 (65536 sequences), 32 rows per workgroup
  lstm1_kernel<<<B2 / 32, 256, 0, stream>>>(cond1, cond2, wih1h, whh1h, bsum1, h1out);

  // cond linear + avg, then bn1 stats
  condlin_kernel<<<B1 / 16, 128, 0, stream>>>(h1out, condwh, cond_b, loavg);
  colstats_kernel<<<HIDN, 256, 0, stream>>>(loavg, B1, HIDN, m1, v1);

  // sample encoder -> feats cols [271,527); operators + bn1(lo) + pad
  sample_kernel<<<B1 / 16, 128, 0, stream>>>(samples, sampwh, samp_b, masks, feats);
  feats_kernel<<<nb((long)B1 * FPAD, 256), 256, 0, stream>>>(
      operators, loavg, m1, v1, bn1_g, bn1_b, feats);

  // lstm2: bottom-up tree recurrence, 16 levels
  for (int s = 0; s < 16; ++s) {
    int lev = 15 - s;
    int w = s & 1, rd = 1 - w;
    lstm2_kernel<<<NN / 16, 256, 0, stream>>>(
        feats, lev, (s == 0) ? 1 : 0, mapping,
        h16[rd], c32[rd], wih2h, whh2h, bsum2,
        h16[w], h32[w], c32[w]);
  }
  float* hfinal = h32[1];   // s=15 wrote buffer 1

  // heads
  colstats_kernel<<<HH, 256, 0, stream>>>(hfinal, NN, HH, m2, v2);
  bnapply_kernel<<<nb((long)NN * HH, 256), 256, 0, stream>>>(
      hfinal, m2, v2, bn2_g, bn2_b, zbuf, (long)NN * HH, HH);

  // head 1 -> o1 @ [0], task1 @ [4096],[6144]
  gemm_relu_kernel<<<NN / 16, 128, 0, stream>>>(zbuf, t1w2h, t1_b2, y1);
  colstats_kernel<<<HIDN, 256, 0, stream>>>(y1, NN, HIDN, m3, v3);
  bnapply_kernel<<<nb((long)NN * HIDN, 256), 256, 0, stream>>>(
      y1, m3, v3, bn3_g, bn3_b, y1n, (long)NN * HIDN, HIDN);
  gemm_relu_kernel<<<NN / 16, 128, 0, stream>>>(y1n, t1w3h, t1_b3, y2);
  headout_kernel<<<nb(NN, 256), 256, 0, stream>>>(y2, t1_Wo, t1_bo, out, 0, 4096, 6144);

  // head 2 -> o2 @ [2048], task2 @ [8192],[10240]
  gemm_relu_kernel<<<NN / 16, 128, 0, stream>>>(zbuf, t2w2h, t2_b2, y1);
  colstats_kernel<<<HIDN, 256, 0, stream>>>(y1, NN, HIDN, m3, v3);
  bnapply_kernel<<<nb((long)NN * HIDN, 256), 256, 0, stream>>>(
      y1, m3, v3, bn3_g, bn3_b, y1n, (long)NN * HIDN, HIDN);
  gemm_relu_kernel<<<NN / 16, 128, 0, stream>>>(y1n, t2w3h, t2_b3, y2);
  headout_kernel<<<nb(NN, 256), 256, 0, stream>>>(y2, t2_Wo, t2_bo, out, 2048, 8192, 10240);
}
